// TTSModel_44066364457056
// MI455X (gfx1250) — matrix operand order repaired and checked by reading, not compiled, not run on hardware
//
#include <hip/hip_runtime.h>
#include <hip/hip_bf16.h>
#include <stdint.h>

// MI455X (gfx1250): all GEMMs via v_wmma_f32_16x16x32_bf16 (bf16 in, f32 acc),
// weights converted once per call to bf16 (L2-resident), recurrent LSTM as
// persistent workgroups (one per direction) with h/c/gates in LDS and the
// x_t activation slab double-buffered via GLOBAL_LOAD_ASYNC_TO_LDS_B128.

typedef __bf16 bf16;
typedef __bf16 v16bf __attribute__((ext_vector_type(16)));
typedef float  v8f   __attribute__((ext_vector_type(8)));

#define B_SZ   32
#define T_SZ   2048
#define EMB    128
#define CDIM   512
#define HDIM   512
#define GDIM   2048           // 4*HDIM
#define KCAT   1024           // HDIM + CDIM
#define M_ALL  (B_SZ * T_SZ)  // 65536
#define NMELS  80

// A-fragment K index pattern for 16-bit WMMA (16x32 A, wave32):
// lanes 0-15 hold rows M=0..15 with K = {0..7, 16..23}; lanes 16-31 same rows,
// K = {8..15, 24..31}.
__device__ __forceinline__ int akidx(int e, int half) {
    return (e < 8) ? (half * 8 + e) : (e + 8 + half * 8);
}

__device__ __forceinline__ float sigm(float x) { return 1.0f / (1.0f + __expf(-x)); }

#define WMMA_BF16(acc, a, b) \
    (acc) = __builtin_amdgcn_wmma_f32_16x16x32_bf16(false, (a), false, (b), (short)0, (acc), false, false)

// ---- gfx1250 async global->LDS copy (ASYNCcnt-tracked; see CDNA5 ISA) ----
__device__ __forceinline__ void async_ld_b128(uint32_t lds_byte_off, const void* gaddr) {
    asm volatile("global_load_async_to_lds_b128 %0, %1, off"
                 :: "v"(lds_byte_off), "v"((uint64_t)(uintptr_t)gaddr)
                 : "memory");
}
__device__ __forceinline__ void wait_async0() {
    asm volatile("s_wait_asynccnt 0x0" ::: "memory");
}

// ---------------- weight conversion kernels ----------------

__global__ void k_cvt_copy(const float* __restrict__ src, bf16* __restrict__ dst, int n) {
    for (int i = blockIdx.x * blockDim.x + threadIdx.x; i < n; i += gridDim.x * blockDim.x)
        dst[i] = (bf16)src[i];
}

// conv_w [512][128][3] (O,I,K) -> B [3][128][512] (tap, in, out)
__global__ void k_cvt_conv(const float* __restrict__ w, bf16* __restrict__ B) {
    const int n = 3 * EMB * CDIM;
    for (int i = blockIdx.x * blockDim.x + threadIdx.x; i < n; i += gridDim.x * blockDim.x) {
        int tap = i / (EMB * CDIM);
        int rem = i % (EMB * CDIM);
        int ci  = rem / CDIM;
        int o   = rem % CDIM;
        B[i] = (bf16)w[(o * EMB + ci) * 3 + tap];
    }
}

// Concat recurrence weights: B[k][g], k<512 from w_hh[g][k], else w_ih[g][k-512]
__global__ void k_cvt_wcat(const float* __restrict__ w_hh, const float* __restrict__ w_ih,
                           bf16* __restrict__ B) {
    const int n = KCAT * GDIM;
    for (int i = blockIdx.x * blockDim.x + threadIdx.x; i < n; i += gridDim.x * blockDim.x) {
        int k = i / GDIM, g = i % GDIM;
        float v = (k < HDIM) ? w_hh[g * HDIM + k] : w_ih[g * CDIM + (k - HDIM)];
        B[i] = (bf16)v;
    }
}

// lin_w [80][1024] -> B[k][n] [1024][80]
__global__ void k_cvt_lin(const float* __restrict__ w, bf16* __restrict__ B) {
    const int n = 2 * HDIM * NMELS;
    for (int i = blockIdx.x * blockDim.x + threadIdx.x; i < n; i += gridDim.x * blockDim.x) {
        int k = i / NMELS, c = i % NMELS;
        B[i] = (bf16)w[c * (2 * HDIM) + k];
    }
}

__global__ void k_zero(float* __restrict__ p, int n) {
    for (int i = blockIdx.x * blockDim.x + threadIdx.x; i < n; i += gridDim.x * blockDim.x)
        p[i] = 0.0f;
}

// ---------------- embedding GEMM: [65536,128] @ [128,128] -> bf16 ----------------

__global__ void k_embed(const float* __restrict__ xin, const bf16* __restrict__ embB,
                        bf16* __restrict__ xemb) {
    const int tilesN = EMB / 16;                       // 8
    int wid = blockIdx.x * (blockDim.x >> 5) + (threadIdx.x >> 5);
    if (wid >= (M_ALL / 16) * tilesN) return;
    int m0 = (wid / tilesN) * 16, n0 = (wid % tilesN) * 16;
    int lane = threadIdx.x & 31, half = lane >> 4, lcol = lane & 15;
    int row = m0 + lcol;

    v8f acc = {0.f, 0.f, 0.f, 0.f, 0.f, 0.f, 0.f, 0.f};
    for (int k0 = 0; k0 < EMB; k0 += 32) {
        v16bf a, b;
#pragma unroll
        for (int e = 0; e < 16; ++e) {
            a[e] = (bf16)xin[(size_t)row * EMB + k0 + akidx(e, half)];
            b[e] = embB[(size_t)(k0 + akidx(e, half)) * EMB + n0 + lcol];
        }
        WMMA_BF16(acc, a, b);
    }
#pragma unroll
    for (int r = 0; r < 8; ++r)
        xemb[(size_t)(m0 + r + 8 * half) * EMB + n0 + lcol] = (bf16)acc[r];
}

// ------------- conv(k=3,pad=1) as 3 shifted GEMMs + ReLU + BN stats -------------

__global__ void k_conv(const bf16* __restrict__ xemb, const bf16* __restrict__ convB,
                       bf16* __restrict__ xc, float* __restrict__ sum,
                       float* __restrict__ sumsq) {
    const int tilesN = CDIM / 16;                      // 32
    int wid = blockIdx.x * (blockDim.x >> 5) + (threadIdx.x >> 5);
    if (wid >= (M_ALL / 16) * tilesN) return;
    int m0 = (wid / tilesN) * 16, n0 = (wid % tilesN) * 16;
    int lane = threadIdx.x & 31, half = lane >> 4, lcol = lane & 15;
    int row = m0 + lcol;
    int t = row & (T_SZ - 1);

    v8f acc = {0.f, 0.f, 0.f, 0.f, 0.f, 0.f, 0.f, 0.f};
    for (int tap = 0; tap < 3; ++tap) {
        int sh = tap - 1;
        bool valid = (t + sh >= 0) && (t + sh < T_SZ);
        const bf16* Bt = convB + (size_t)tap * EMB * CDIM;
        const bf16* arow = xemb + (size_t)(row + sh) * EMB;
        for (int k0 = 0; k0 < EMB; k0 += 32) {
            v16bf a, b;
#pragma unroll
            for (int e = 0; e < 16; ++e) {
                a[e] = valid ? arow[k0 + akidx(e, half)] : (bf16)0.0f;
                b[e] = Bt[(size_t)(k0 + akidx(e, half)) * CDIM + n0 + lcol];
            }
            WMMA_BF16(acc, a, b);
        }
    }
    float s1 = 0.f, s2 = 0.f;
#pragma unroll
    for (int r = 0; r < 8; ++r) {
        float v = fmaxf(acc[r], 0.0f);
        xc[(size_t)(m0 + r + 8 * half) * CDIM + n0 + lcol] = (bf16)v;
        s1 += v; s2 += v * v;
    }
    atomicAdd(&sum[n0 + lcol], s1);
    atomicAdd(&sumsq[n0 + lcol], s2);
}

__global__ void k_bnstats(const float* __restrict__ sum, const float* __restrict__ sumsq,
                          const float* __restrict__ gamma, const float* __restrict__ beta,
                          float* __restrict__ scale, float* __restrict__ shift) {
    int c = blockIdx.x * blockDim.x + threadIdx.x;
    if (c < CDIM) {
        const float invN = 1.0f / (float)M_ALL;
        float m = sum[c] * invN;
        float v = sumsq[c] * invN - m * m;
        float sc = gamma[c] * rsqrtf(v + 1e-5f);
        scale[c] = sc;
        shift[c] = beta[c] - m * sc;
    }
}

__global__ void k_bnapply(bf16* __restrict__ xc, const float* __restrict__ scale,
                          const float* __restrict__ shift) {
    size_t n = (size_t)M_ALL * CDIM;
    for (size_t i = blockIdx.x * (size_t)blockDim.x + threadIdx.x; i < n;
         i += (size_t)gridDim.x * blockDim.x) {
        int c = (int)(i & (CDIM - 1));
        xc[i] = (bf16)((float)xc[i] * scale[c] + shift[c]);
    }
}

// ------------- persistent bidirectional LSTM: one workgroup per direction -------------
// Per step: g[32,2048] = [h | x_t][32,1024] @ Wcat[1024,2048] + b  (WMMA bf16)
// LDS: h bf16 32KB + c f32 64KB + g bf16 128KB + x double-buffer 2x32KB = 288KB.
// x_{t+1} slab is prefetched with global_load_async_to_lds_b128 (ASYNCcnt) while
// step t's gate GEMM runs on WMMA; s_wait_asynccnt 0 before the step barrier.

__global__ void __launch_bounds__(1024) k_lstm(
    const bf16* __restrict__ xc, const bf16* __restrict__ wF, const bf16* __restrict__ wR,
    const float* __restrict__ bF, const float* __restrict__ bR,
    bf16* __restrict__ hfOut, bf16* __restrict__ hbOut) {
    extern __shared__ char smem[];
    bf16*  h_lds = (bf16*)smem;                                  // [32][512]   32KB
    float* c_lds = (float*)(smem + 32768);                       // [32][512]   64KB
    bf16*  g_lds = (bf16*)(smem + 32768 + 65536);                // [32][2048] 128KB
    bf16*  xbuf  = (bf16*)(smem + 32768 + 65536 + 131072);       // [2][32][512] 64KB

    int dir = blockIdx.x;
    const bf16*  W    = dir ? wR : wF;
    const float* bias = dir ? bR : bF;
    bf16*        hout = dir ? hbOut : hfOut;

    int tid = threadIdx.x;
    for (int i = tid; i < B_SZ * HDIM; i += 1024) { h_lds[i] = (bf16)0.0f; c_lds[i] = 0.0f; }

    int lane = tid & 31, half = lane >> 4, lcol = lane & 15;
    int wv = tid >> 5;  // wave 0..31

    // Gate bias is time-invariant: cache each wave's 8 tile biases in registers.
    float biasr[8];
#pragma unroll
    for (int i = 0; i < 8; ++i) {
        int n0 = ((wv * 8 + i) & 127) * 16;
        biasr[i] = bias[n0 + lcol];
    }

    // Prologue: async-stage the first x slab into xbuf[0]. Each thread moves
    // 2 x 16B of the 32KB slab.
    {
        int t0 = dir ? (T_SZ - 1) : 0;
#pragma unroll
        for (int i = 0; i < 2; ++i) {
            int fb = tid * 32 + i * 16;                 // byte index into slab
            int bb = fb >> 10;                          // 1024 bytes per batch row
            int bc = fb & 1023;
            const char* g = (const char*)xc + ((size_t)(bb * T_SZ + t0) * CDIM) * 2 + bc;
            async_ld_b128((uint32_t)(uintptr_t)((char*)xbuf + fb), g);
        }
        wait_async0();
    }
    __syncthreads();

    for (int s = 0; s < T_SZ; ++s) {
        int t = dir ? (T_SZ - 1 - s) : s;
        int cur = s & 1, nxt = cur ^ 1;

        // Kick off async staging of the next step's x slab (overlaps the GEMM).
        {
            int tn = dir ? (T_SZ - 2 - s) : (s + 1);
            tn = (tn < 0) ? 0 : ((tn > T_SZ - 1) ? T_SZ - 1 : tn);
#pragma unroll
            for (int i = 0; i < 2; ++i) {
                int fb = tid * 32 + i * 16;
                int bb = fb >> 10;
                int bc = fb & 1023;
                const char* g = (const char*)xc + ((size_t)(bb * T_SZ + tn) * CDIM) * 2 + bc;
                async_ld_b128((uint32_t)(uintptr_t)((char*)xbuf + nxt * 32768 + fb), g);
            }
        }

        const bf16* xcur = xbuf + cur * (B_SZ * HDIM);

        // ---- gate GEMM: 256 tiles (2 x 128), 8 per wave, K=1024 (32 wmma each)
#pragma unroll 1
        for (int i = 0; i < 8; ++i) {
            int tt = wv * 8 + i;
            int mt = tt >> 7, nt = tt & 127;
            int m0 = mt * 16, n0 = nt * 16;
            int row = m0 + lcol;  // batch index
            float bv = biasr[i];
            v8f acc = {bv, bv, bv, bv, bv, bv, bv, bv};
#pragma unroll 1
            for (int k0 = 0; k0 < KCAT; k0 += 32) {
                v16bf a, b;
#pragma unroll
                for (int e = 0; e < 16; ++e) {
                    int ke = k0 + akidx(e, half);
                    a[e] = (ke < HDIM) ? h_lds[row * HDIM + ke] : xcur[row * HDIM + (ke - HDIM)];
                    b[e] = W[(size_t)ke * GDIM + n0 + lcol];
                }
                if (k0 + 32 < KCAT)  // keep next weight slab hot (global_prefetch_b8)
                    __builtin_prefetch(&W[(size_t)(k0 + 32 + lane) * GDIM + n0], 0, 1);
                WMMA_BF16(acc, a, b);
            }
#pragma unroll
            for (int r = 0; r < 8; ++r)
                g_lds[(m0 + r + 8 * half) * GDIM + n0 + lcol] = (bf16)acc[r];
        }
        __syncthreads();

        // ---- activations, state update, write h to global
        for (int idx = tid; idx < B_SZ * HDIM; idx += 1024) {
            int bb = idx >> 9, j = idx & (HDIM - 1);
            const bf16* gr = g_lds + bb * GDIM;
            float gi = (float)gr[j];
            float gf = (float)gr[HDIM + j];
            float gg = (float)gr[2 * HDIM + j];
            float go = (float)gr[3 * HDIM + j];
            float c = c_lds[idx];
            c = sigm(gf) * c + sigm(gi) * tanhf(gg);
            float h = sigm(go) * tanhf(c);
            c_lds[idx] = c;
            h_lds[idx] = (bf16)h;
            hout[((size_t)bb * T_SZ + t) * HDIM + j] = (bf16)h;
        }
        wait_async0();   // next x slab resident before anyone reads it next step
        __syncthreads();
    }
}

// ------------- output linear: [65536,1024] @ [1024,80] + b -> f32 -------------

__global__ void k_linear(const bf16* __restrict__ hf, const bf16* __restrict__ hb,
                         const bf16* __restrict__ linB, const float* __restrict__ lin_b,
                         float* __restrict__ out) {
    const int tilesN = NMELS / 16;                     // 5
    int wid = blockIdx.x * (blockDim.x >> 5) + (threadIdx.x >> 5);
    if (wid >= (M_ALL / 16) * tilesN) return;
    int m0 = (wid / tilesN) * 16, n0 = (wid % tilesN) * 16;
    int lane = threadIdx.x & 31, half = lane >> 4, lcol = lane & 15;
    int row = m0 + lcol;

    float bv = lin_b[n0 + lcol];
    v8f acc = {bv, bv, bv, bv, bv, bv, bv, bv};
    for (int k0 = 0; k0 < 2 * HDIM; k0 += 32) {
        v16bf a, b;
#pragma unroll
        for (int e = 0; e < 16; ++e) {
            int ke = k0 + akidx(e, half);
            a[e] = (ke < HDIM) ? hf[(size_t)row * HDIM + ke]
                               : hb[(size_t)row * HDIM + (ke - HDIM)];
            b[e] = linB[(size_t)ke * NMELS + n0 + lcol];
        }
        WMMA_BF16(acc, a, b);
    }
#pragma unroll
    for (int r = 0; r < 8; ++r)
        out[(size_t)(m0 + r + 8 * half) * NMELS + n0 + lcol] = acc[r];
}

// ---------------------------------------------------------------------------

extern "C" void kernel_launch(void* const* d_in, const int* in_sizes, int n_in,
                              void* d_out, int out_size, void* d_ws, size_t ws_size,
                              hipStream_t stream) {
    const float* x_in   = (const float*)d_in[0];
    const float* emb    = (const float*)d_in[1];
    const float* conv_w = (const float*)d_in[2];
    const float* gamma  = (const float*)d_in[3];
    const float* beta   = (const float*)d_in[4];
    const float* w_ih_f = (const float*)d_in[5];
    const float* w_hh_f = (const float*)d_in[6];
    const float* b_f    = (const float*)d_in[7];
    const float* w_ih_b = (const float*)d_in[8];
    const float* w_hh_b = (const float*)d_in[9];
    const float* b_b    = (const float*)d_in[10];
    const float* lin_w  = (const float*)d_in[11];
    const float* lin_b  = (const float*)d_in[12];
    float* out = (float*)d_out;
    (void)in_sizes; (void)n_in; (void)out_size; (void)ws_size;

    char* ws = (char*)d_ws;
    size_t off = 0;
    auto take = [&](size_t bytes) -> char* {
        char* p = ws + off;
        off = (off + bytes + 255) & ~(size_t)255;
        return p;
    };
    bf16*  embB  = (bf16*)take((size_t)EMB * EMB * 2);
    bf16*  convB = (bf16*)take((size_t)3 * EMB * CDIM * 2);
    bf16*  wcatF = (bf16*)take((size_t)KCAT * GDIM * 2);
    bf16*  wcatR = (bf16*)take((size_t)KCAT * GDIM * 2);
    bf16*  linB  = (bf16*)take((size_t)KCAT * NMELS * 2);
    bf16*  xemb  = (bf16*)take((size_t)M_ALL * EMB * 2);
    bf16*  xcb   = (bf16*)take((size_t)M_ALL * CDIM * 2);
    bf16*  hf    = (bf16*)take((size_t)M_ALL * HDIM * 2);
    bf16*  hb    = (bf16*)take((size_t)M_ALL * HDIM * 2);
    float* sum   = (float*)take(CDIM * 4);
    float* sumsq = (float*)take(CDIM * 4);
    float* scale = (float*)take(CDIM * 4);
    float* shift = (float*)take(CDIM * 4);

    // 1) weight conversion to bf16 (L2-resident working set)
    k_cvt_copy<<<64, 256, 0, stream>>>(emb, embB, EMB * EMB);
    k_cvt_conv<<<768, 256, 0, stream>>>(conv_w, convB);
    k_cvt_wcat<<<4096, 256, 0, stream>>>(w_hh_f, w_ih_f, wcatF);
    k_cvt_wcat<<<4096, 256, 0, stream>>>(w_hh_b, w_ih_b, wcatR);
    k_cvt_lin<<<320, 256, 0, stream>>>(lin_w, linB);
    k_zero<<<2, 256, 0, stream>>>(sum, CDIM);
    k_zero<<<2, 256, 0, stream>>>(sumsq, CDIM);

    // 2) embedding GEMM  (tiles = 4096*8, 8 waves/block)
    k_embed<<<4096, 256, 0, stream>>>(x_in, embB, xemb);
    // 3) conv + relu + BN stats (tiles = 4096*32)
    k_conv<<<16384, 256, 0, stream>>>(xemb, convB, xcb, sum, sumsq);
    k_bnstats<<<2, 256, 0, stream>>>(sum, sumsq, gamma, beta, scale, shift);
    k_bnapply<<<4096, 256, 0, stream>>>(xcb, scale, shift);

    // 4) persistent bidirectional LSTM: 2 workgroups, 288KB dynamic LDS each
    size_t lstm_lds = 32768 + 65536 + 131072 + 65536;
    k_lstm<<<2, 1024, lstm_lds, stream>>>(xcb, wcatF, wcatR, b_f, b_b, hf, hb);

    // 5) output projection (tiles = 4096*5)
    k_linear<<<2560, 256, 0, stream>>>(hf, hb, linB, lin_b, out);
}
